// EncoderLayer_86337432584213
// MI455X (gfx1250) — compile-verified, matrix operand order checked by reading, not run on hardware
//
#include <hip/hip_runtime.h>
#include <math.h>

#define B_ 4
#define N_ 256
#define D_ 256
#define H_ 8
#define E_ 32
#define FF_ 512
#define DK_ 32
#define SCALE_ 0.17677669529663687f   // 32^-0.5
#define NEG_ (-9.0e15f)

typedef __attribute__((ext_vector_type(16))) _Float16 v16h;
typedef __attribute__((ext_vector_type(8)))  _Float16 v8h;
typedef __attribute__((ext_vector_type(8)))  float    v8f;
typedef __attribute__((ext_vector_type(4)))  unsigned int u32x4;
typedef __attribute__((ext_vector_type(8)))  int i32x8;
typedef __attribute__((ext_vector_type(4)))  int i32x4;

// ---------------------------------------------------------------------------
// WMMA helpers (CDNA5 wave32, v_wmma_f32_16x16x32_f16)
// A 16x32 f16: lane<16 -> row=lane, K = {k0..k0+7, k0+16..k0+23}
//              lane>=16 -> row=lane-16, K = {k0+8..k0+15, k0+24..k0+31}
// B 32x16 f16: identical striping with n as the per-lane index, so a
//              B-fragment of W is an A-fragment of W^T  ->  all weights are
//              stored transposed and every fragment load is 2x b128.
// C 16x16 f32: lane = column (mod 16), 8 rows per lane (+8 for lane>=16)
// ---------------------------------------------------------------------------
__device__ __forceinline__ v8f wmma16(v16h a, v16h b, v8f c) {
  return __builtin_amdgcn_wmma_f32_16x16x32_f16(false, a, false, b, (short)0, c,
                                                false, false);
}

// Fragment load: two 16B vector loads (all call sites are 16B aligned:
// ld is a multiple of 8 halves with 16B-aligned rows, kb*2 % 16 == 0).
__device__ __forceinline__ v16h load_f16(const _Float16* A, int ld, long m0,
                                         int k0, int lane) {
  long row = m0 + (lane & 15);
  int kb = k0 + ((lane >> 4) << 3);
  const _Float16* p = A + row * (long)ld + kb;
  v8h lo = *(const v8h*)p;
  v8h hi = *(const v8h*)(p + 16);
  v16h a;
#pragma unroll
  for (int i = 0; i < 8; ++i) { a[i] = lo[i]; a[i + 8] = hi[i]; }
  return a;
}

__device__ __forceinline__ float gelu_exact(float v) {
  return 0.5f * v * (1.0f + erff(v * 0.70710678118654752f));
}

// ---------------------------------------------------------------------------
// Tensor Data Mover: 2D tile global->LDS with LDS row padding (bank skew +
// 16B row alignment). dim0 = contiguous length (elements), dim1 = rows,
// 2-byte elements. pad: insert (pad_amount_code+1) dwords every
// 2^(pad_interval_code+1) dwords.
// ---------------------------------------------------------------------------
#if __has_builtin(__builtin_amdgcn_tensor_load_to_lds)
#define HAVE_TDM 1
__device__ __forceinline__ void tdm_load_2d_f16(const void* gptr, unsigned lds_off,
                                                unsigned dim0, unsigned dim1,
                                                unsigned pad_interval,
                                                unsigned pad_amount) {
  unsigned long long ga = (unsigned long long)(size_t)gptr;
  u32x4 g0;
  g0[0] = 1u;                                    // count=1, user descriptor
  g0[1] = lds_off;                               // lds_addr [63:32]
  g0[2] = (unsigned)(ga & 0xffffffffu);          // global_addr lo
  g0[3] = (unsigned)((ga >> 32) & 0x01ffffffu) | (2u << 30);  // addr hi | type=2
  i32x8 g1;
  g1[0] = (int)((1u << 16) | (1u << 20) | (pad_interval << 22) | (pad_amount << 25));
  g1[1] = (int)((dim0 & 0xffffu) << 16);                  // tensor_dim0 lo16
  g1[2] = (int)((dim0 >> 16) | ((dim1 & 0xffffu) << 16)); // dim0 hi | dim1 lo
  g1[3] = (int)((dim1 >> 16) | ((dim0 & 0xffffu) << 16)); // dim1 hi | tile_dim0
  g1[4] = (int)(dim1 & 0xffffu);                          // tile_dim1 (tile_dim2=0)
  g1[5] = (int)dim0;                                      // tensor_dim0_stride lo32
  g1[6] = 0;
  g1[7] = 0;
  i32x4 z4 = {0, 0, 0, 0};
#if __clang_major__ >= 23
  i32x8 z8 = {0, 0, 0, 0, 0, 0, 0, 0};
  __builtin_amdgcn_tensor_load_to_lds(g0, g1, z4, z4, z8, 0);
#else
  __builtin_amdgcn_tensor_load_to_lds(g0, g1, z4, z4, 0);
#endif
}
#else
#define HAVE_TDM 0
#endif

// ---------------------------------------------------------------------------
// fp32 [K][Nn] -> fp16 transposed [Nn][K] (one-time weight prep)
// ---------------------------------------------------------------------------
__global__ void k_cvt_t(const float* __restrict__ s, _Float16* __restrict__ d,
                        int K, int Nn) {
  int i = blockIdx.x * 256 + threadIdx.x;
  if (i < K * Nn) {
    int k = i / Nn, n = i % Nn;
    d[(long)n * K + k] = (_Float16)s[i];
  }
}

// ---------------------------------------------------------------------------
// Graph norm over node axis (dim=1): per (b,d) stats. Coalesced over d.
// ---------------------------------------------------------------------------
__global__ void k_graph_norm(const float* __restrict__ t, const float* __restrict__ w,
                             const float* __restrict__ bb, const float* __restrict__ ms,
                             _Float16* __restrict__ yh) {
  int b = blockIdx.x;
  int d = blockIdx.y * 64 + threadIdx.x;
  const float* p = t + (long)b * N_ * D_ + d;
  float s = 0.f;
  for (int n = 0; n < N_; ++n) s += p[(long)n * D_];
  float m = ms[d] * (s * (1.0f / N_));
  float v = 0.f;
  for (int n = 0; n < N_; ++n) { float u = p[(long)n * D_] - m; v += u * u; }
  float inv = 1.0f / (sqrtf(v * (1.0f / N_)) + 1e-6f);
  float ww = w[d], bv = bb[d];
  for (int n = 0; n < N_; ++n)
    yh[((long)b * N_ + n) * D_ + d] = (_Float16)(ww * (p[(long)n * D_] - m) * inv + bv);
}

// ---------------------------------------------------------------------------
// QKV: [1024,256] @ W^T, q scaled. q,k head-major [B,H,N,DK]; v transposed
// [B,H,DK,N] so that P@V B-fragments are vector loads.
// ---------------------------------------------------------------------------
__global__ void k_qkv(const _Float16* __restrict__ yh,
                      const _Float16* WqT, const float* bq,
                      const _Float16* WkT, const float* bk,
                      const _Float16* WvT, const float* bv,
                      _Float16* qh, _Float16* kh, _Float16* vt) {
  int lane = threadIdx.x;
  long m0 = (long)blockIdx.x * 16;
  int n0 = blockIdx.y * 16;
  int which = blockIdx.z;
  const _Float16* WT = which == 0 ? WqT : (which == 1 ? WkT : WvT);
  const float* bias = which == 0 ? bq : (which == 1 ? bk : bv);
  v8f c = {};
  for (int k0 = 0; k0 < D_; k0 += 32) {
    v16h a = load_f16(yh, D_, m0, k0, lane);
    v16h bm = load_f16(WT, D_, n0, k0, lane);
    c = wmma16(a, bm, c);
  }
  int col = n0 + (lane & 15);
  long rb = m0 + ((lane >> 4) << 3);
  float scale = (which == 0) ? SCALE_ : 1.0f;
  float bvv = bias[col];
  int h = col >> 5, dk = col & 31;
#pragma unroll
  for (int r = 0; r < 8; ++r) {
    long row = rb + r;
    long b = row >> 8; long nn = row & 255;
    _Float16 val = (_Float16)((c[r] + bvv) * scale);
    if (which == 2)
      vt[((b * H_ + h) * DK_ + dk) * N_ + nn] = val;
    else if (which == 1)
      kh[((b * H_ + h) * N_ + nn) * DK_ + dk] = val;
    else
      qh[((b * H_ + h) * N_ + nn) * DK_ + dk] = val;
  }
}

// ---------------------------------------------------------------------------
// scores[bh,i,j] = Q_i . K_j  (K=DK=32, one WMMA), + adjacency mask
// ---------------------------------------------------------------------------
__global__ void k_scores(const _Float16* __restrict__ qh, const _Float16* __restrict__ kh,
                         const int* __restrict__ adj, const int* __restrict__ use_adj_p,
                         float* __restrict__ scores) {
  int lane = threadIdx.x;
  int i0 = blockIdx.x * 16, j0 = blockIdx.y * 16;
  int bh = blockIdx.z; int b = bh >> 3;
  const _Float16* Q = qh + (long)bh * N_ * DK_;
  const _Float16* K = kh + (long)bh * N_ * DK_;
  v16h a = load_f16(Q, DK_, i0, 0, lane);
  v16h bm = load_f16(K, DK_, j0, 0, lane);  // B = K^T: fragment of K rows
  v8f c = {};
  c = wmma16(a, bm, c);
  int ua = *use_adj_p;
  int col = j0 + (lane & 15);
  int rb = i0 + ((lane >> 4) << 3);
#pragma unroll
  for (int r = 0; r < 8; ++r) {
    int row = rb + r;
    float s = c[r];
    if (ua) {
      int av = adj[((long)b * N_ + row) * N_ + col];
      s = (av > 0 ? s : NEG_) * (float)av;
    }
    scores[((long)bh * N_ + row) * N_ + col] = s;
  }
}

// ---------------------------------------------------------------------------
// LayerNorm over E=32 per edge row (one wave per row).
// Optionally also e_n @ Wep + bep  ->  edge_bias[b,h,n2,n1]  (swapaxes(1,3)).
// ---------------------------------------------------------------------------
__global__ void k_edge_ln(const float* __restrict__ ein, const float* __restrict__ lw,
                          const float* __restrict__ lb, _Float16* __restrict__ eout,
                          const float* __restrict__ Wep, const float* __restrict__ bep,
                          float* __restrict__ ebias) {
  long row = (long)blockIdx.x * (blockDim.x >> 5) + (threadIdx.x >> 5);
  int lane = threadIdx.x & 31;
  float v = ein[row * E_ + lane];
  float s = v;
#pragma unroll
  for (int o = 16; o > 0; o >>= 1) s += __shfl_xor(s, o, 32);
  float mu = s * (1.0f / E_);
  float d = v - mu;
  float q = d * d;
#pragma unroll
  for (int o = 16; o > 0; o >>= 1) q += __shfl_xor(q, o, 32);
  float en = lw[lane] * d * rsqrtf(q * (1.0f / E_) + 1e-5f) + lb[lane];
  eout[row * E_ + lane] = (_Float16)en;
  if (ebias) {
    long b = row >> 16;
    int n1 = (int)((row >> 8) & 255);
    int n2 = (int)(row & 255);
#pragma unroll
    for (int h = 0; h < H_; ++h) {
      float t = en * Wep[lane * H_ + h];
#pragma unroll
      for (int o = 16; o > 0; o >>= 1) t += __shfl_xor(t, o, 32);
      if (lane == h)  // edge_bias[b,h,i,j] = (e_n@Wep)[b,j,i,h] + bep
        ebias[((b * H_ + h) * N_ + n2) * N_ + n1] = t + bep[h];
    }
  }
}

// ---------------------------------------------------------------------------
// Dual softmax over j: P = softmax(s*attn_bias) + softmax(s*edge_bias*adj1)
// ---------------------------------------------------------------------------
__global__ void k_softmax_dual(const float* __restrict__ scores,
                               const float* __restrict__ attn_bias,
                               const float* __restrict__ edge_bias,
                               const int* __restrict__ adj1,
                               _Float16* __restrict__ P) {
  __shared__ float red[256];
  int row = blockIdx.x;  // bh*N + i
  int bh = row >> 8; int i = row & 255; int b = bh >> 3;
  int j = threadIdx.x;
  long off = (long)row * N_ + j;
  float s = scores[off];
  float g = s * attn_bias[off];
  float l = s * edge_bias[off] * (float)adj1[((long)b * N_ + i) * N_ + j];

  red[j] = g; __syncthreads();
  for (int st = 128; st > 0; st >>= 1) { if (j < st) red[j] = fmaxf(red[j], red[j + st]); __syncthreads(); }
  float gm = red[0]; __syncthreads();
  float ge = __expf(g - gm);
  red[j] = ge; __syncthreads();
  for (int st = 128; st > 0; st >>= 1) { if (j < st) red[j] += red[j + st]; __syncthreads(); }
  float gs = red[0]; __syncthreads();

  red[j] = l; __syncthreads();
  for (int st = 128; st > 0; st >>= 1) { if (j < st) red[j] = fmaxf(red[j], red[j + st]); __syncthreads(); }
  float lm = red[0]; __syncthreads();
  float le = __expf(l - lm);
  red[j] = le; __syncthreads();
  for (int st = 128; st > 0; st >>= 1) { if (j < st) red[j] += red[j + st]; __syncthreads(); }
  float ls = red[0];

  P[off] = (_Float16)(ge / gs + le / ls);
}

// ---------------------------------------------------------------------------
// out_local+out_long concatenated heads: attn_cat[b,i,h*32+dk] = P @ V
// V is stored transposed [B,H,DK,N] -> B-fragments are vector loads.
// ---------------------------------------------------------------------------
__global__ void k_attn_v(const _Float16* __restrict__ P, const _Float16* __restrict__ vt,
                         _Float16* __restrict__ attn_cat) {
  int lane = threadIdx.x;
  int i0 = blockIdx.x * 16;
  int bh = blockIdx.y; int b = bh >> 3, h = bh & 7;
  const _Float16* Pp = P + (long)bh * N_ * N_;
  const _Float16* V = vt + (long)bh * DK_ * N_;
  v8f c0 = {}, c1 = {};
  for (int k0 = 0; k0 < N_; k0 += 32) {
    v16h a = load_f16(Pp, N_, i0, k0, lane);
    v16h b0 = load_f16(V, N_, 0, k0, lane);
    v16h b1 = load_f16(V, N_, 16, k0, lane);
    c0 = wmma16(a, b0, c0);
    c1 = wmma16(a, b1, c1);
  }
  int cl = lane & 15;
  int rb = i0 + ((lane >> 4) << 3);
#pragma unroll
  for (int r = 0; r < 8; ++r) {
    long orow = ((long)b * N_ + (rb + r)) * D_ + h * DK_;
    attn_cat[orow + cl] = (_Float16)c0[r];
    attn_cat[orow + 16 + cl] = (_Float16)c1[r];
  }
}

// ---------------------------------------------------------------------------
// x1 = x + attn_cat @ Wo + bo    (M=1024, K=256, N=256, B = WoT fragments)
// ---------------------------------------------------------------------------
__global__ void k_proj_res(const _Float16* __restrict__ Ah, const _Float16* __restrict__ WT,
                           const float* __restrict__ bias, const float* __restrict__ res,
                           float* __restrict__ out) {
  int lane = threadIdx.x;
  long m0 = (long)blockIdx.x * 16;
  int n0 = blockIdx.y * 16;
  v8f c = {};
  for (int k0 = 0; k0 < D_; k0 += 32) {
    v16h a = load_f16(Ah, D_, m0, k0, lane);
    v16h bm = load_f16(WT, D_, n0, k0, lane);
    c = wmma16(a, bm, c);
  }
  int col = n0 + (lane & 15);
  long rb = m0 + ((lane >> 4) << 3);
  float bv = bias[col];
#pragma unroll
  for (int r = 0; r < 8; ++r) {
    long idx = (rb + r) * (long)D_ + col;
    out[idx] = res[idx] + c[r] + bv;
  }
}

// ---------------------------------------------------------------------------
// t[b,n1,n2,:] = e_n + (sum_h scores[b,h,n2,n1]*Wap[h,:] + bap) * adj1[b,n1,n2]
// ---------------------------------------------------------------------------
__global__ void k_edge_fuse(const float* __restrict__ scores, const float* __restrict__ Wap,
                            const float* __restrict__ bap, const int* __restrict__ adj1,
                            _Float16* __restrict__ en_t) {
  long row = (long)blockIdx.x * (blockDim.x >> 5) + (threadIdx.x >> 5);
  int lane = threadIdx.x & 31;
  long b = row >> 16;
  int n1 = (int)((row >> 8) & 255);
  int n2 = (int)(row & 255);
  float sv = 0.f;
  if (lane < H_) sv = scores[((b * H_ + lane) * N_ + n2) * N_ + n1];
  float xp = bap[lane];
#pragma unroll
  for (int h = 0; h < H_; ++h) xp += __shfl(sv, h, 32) * Wap[h * E_ + lane];
  xp *= (float)adj1[row];
  float t = (float)en_t[row * E_ + lane] + xp;
  en_t[row * E_ + lane] = (_Float16)t;
}

// ---------------------------------------------------------------------------
// e1 = edge_fea + t @ Woe   (M=262144, K=32, N=32, B = WoeT fragments)
// ---------------------------------------------------------------------------
__global__ void k_edge_woe(const _Float16* __restrict__ Th, const _Float16* __restrict__ WoeT,
                           const float* __restrict__ edge_fea, float* __restrict__ e1) {
  int lane = threadIdx.x;
  long m0 = (long)blockIdx.x * 16;
  v16h a = load_f16(Th, E_, m0, 0, lane);
  v16h b0 = load_f16(WoeT, E_, 0, 0, lane);
  v16h b1 = load_f16(WoeT, E_, 16, 0, lane);
  v8f c0 = {}, c1 = {};
  c0 = wmma16(a, b0, c0);
  c1 = wmma16(a, b1, c1);
  int cl = lane & 15;
  long rb = m0 + ((lane >> 4) << 3);
#pragma unroll
  for (int r = 0; r < 8; ++r) {
    long idx = (rb + r) * (long)E_;
    e1[idx + cl] = edge_fea[idx + cl] + c0[r];
    e1[idx + 16 + cl] = edge_fea[idx + 16 + cl] + c1[r];
  }
}

// ---------------------------------------------------------------------------
// Generic fused 2-layer FFN (node path): out = res + gelu(A@W1+b1)@W2+b2.
// Weights given transposed (W1T [FF][K], W2T [Nout][FF]); hidden in LDS.
// ---------------------------------------------------------------------------
__global__ void k_fused_ffn(const _Float16* __restrict__ Ah, int K,
                            const _Float16* __restrict__ W1T, const float* __restrict__ b1,
                            const _Float16* __restrict__ W2T, const float* __restrict__ b2,
                            int Nout, const float* __restrict__ res,
                            float* __restrict__ out) {
  __shared__ _Float16 hid[16 * FF_];
  int lane = threadIdx.x;
  long m0 = (long)blockIdx.x * 16;
  int cl = lane & 15;
  int rsel = ((lane >> 4) << 3);
  for (int nt = 0; nt < FF_; nt += 16) {
    v8f c = {};
    for (int k0 = 0; k0 < K; k0 += 32) {
      v16h a = load_f16(Ah, K, m0, k0, lane);
      v16h bm = load_f16(W1T, K, nt, k0, lane);
      c = wmma16(a, bm, c);
    }
    int col = nt + cl;
    float bv = b1[col];
#pragma unroll
    for (int r = 0; r < 8; ++r)
      hid[(rsel + r) * FF_ + col] = (_Float16)gelu_exact(c[r] + bv);
  }
  __syncthreads();
  for (int nt = 0; nt < Nout; nt += 16) {
    v8f c = {};
    for (int k0 = 0; k0 < FF_; k0 += 32) {
      v16h a = load_f16(hid, FF_, 0, k0, lane);
      v16h bm = load_f16(W2T, FF_, nt, k0, lane);
      c = wmma16(a, bm, c);
    }
    int col = nt + cl;
    float bv = b2[col];
#pragma unroll
    for (int r = 0; r < 8; ++r) {
      long idx = (m0 + rsel + r) * (long)Nout + col;
      out[idx] = res[idx] + c[r] + bv;
    }
  }
}

// ---------------------------------------------------------------------------
// Specialized edge FFN (dominant GEMM): 262144 rows of 32 -> 512 -> 32.
// W1e^T + W2e^T staged once per block into LDS by the Tensor Data Mover with
// row padding (16B-aligned, bank-skewed pitches); 4 waves x 16 rows/block.
// Per 16-row tile: 1 KB of global A traffic, 64 WMMAs, all fragments via
// b128 vector loads (global or ds).
// ---------------------------------------------------------------------------
#define W1T_PITCH 40                        // 32 halves + 16B pad  (80B rows)
#define W2T_PITCH 520                       // 512 halves + 16B pad (1040B rows)
#define HID_PITCH 520
#define W1T_BYTES (FF_ * W1T_PITCH * 2)     // 40960
#define W2T_BYTES (E_ * W2T_PITCH * 2)      // 33280
#define HID_BYTES (16 * HID_PITCH * 2)      // 16640

__global__ void k_edge_ffn(const _Float16* __restrict__ Ah,
                           const _Float16* __restrict__ W1T, const float* __restrict__ b1,
                           const _Float16* __restrict__ W2T, const float* __restrict__ b2,
                           const float* __restrict__ res, float* __restrict__ out) {
  __shared__ __align__(16) char smem[W1T_BYTES + W2T_BYTES + 4 * HID_BYTES];
  _Float16* w1t = (_Float16*)smem;
  _Float16* w2t = (_Float16*)(smem + W1T_BYTES);
  int tid = threadIdx.x;
  int wave = tid >> 5;
  int lane = tid & 31;
  _Float16* hid = (_Float16*)(smem + W1T_BYTES + W2T_BYTES + wave * HID_BYTES);

#if HAVE_TDM
  if (tid < 32) {  // wave 0 drives the TDM; EXEC is ignored by tensor ops
    // W1e^T [512][32]: 64B rows, pad 16B every 64B  -> pitch 80B
    tdm_load_2d_f16(W1T, (unsigned)(size_t)(void*)w1t, E_, FF_, 3, 3);
    // W2e^T [32][512]: 1024B rows, pad 16B every 1024B -> pitch 1040B
    tdm_load_2d_f16(W2T, (unsigned)(size_t)(void*)w2t, FF_, E_, 7, 3);
    __builtin_amdgcn_s_wait_tensorcnt(0);
  }
#else
  for (int i = tid; i < E_ * FF_; i += 128) {
    w1t[(i / E_) * W1T_PITCH + (i % E_)] = W1T[i];
    w2t[(i / FF_) * W2T_PITCH + (i % FF_)] = W2T[i];
  }
#endif
  __syncthreads();

  long m0 = ((long)blockIdx.x * 4 + wave) * 16;
  int cl = lane & 15;
  int rsel = ((lane >> 4) << 3);
  if (lane == 0) __builtin_prefetch(Ah + (m0 + 64) * (long)E_, 0, 1);

  // layer 1: A fragment loaded once (K=32), 32 WMMAs against LDS-resident W1e^T
  v16h a = load_f16(Ah, E_, m0, 0, lane);
#pragma unroll 4
  for (int nt = 0; nt < FF_; nt += 16) {
    v16h bm = load_f16(w1t, W1T_PITCH, nt, 0, lane);
    v8f c = {};
    c = wmma16(a, bm, c);
    int col = nt + cl;
    float bv = b1[col];
#pragma unroll
    for (int r = 0; r < 8; ++r)
      hid[(rsel + r) * HID_PITCH + col] = (_Float16)gelu_exact(c[r] + bv);
  }
  // same-wave LDS ops are in-order; no barrier needed between phases
  // layer 2: out = res + hidden @ W2e + b2 (2 tiles x 16 k-steps)
  for (int nt = 0; nt < E_; nt += 16) {
    v8f c = {};
#pragma unroll 4
    for (int k0 = 0; k0 < FF_; k0 += 32) {
      v16h a2 = load_f16(hid, HID_PITCH, 0, k0, lane);
      v16h bm = load_f16(w2t, W2T_PITCH, nt, k0, lane);
      c = wmma16(a2, bm, c);
    }
    int col = nt + cl;
    float bv = b2[col];
#pragma unroll
    for (int r = 0; r < 8; ++r) {
      long idx = (m0 + rsel + r) * (long)E_ + col;
      out[idx] = res[idx] + c[r] + bv;
    }
  }
}

// ---------------------------------------------------------------------------
extern "C" void kernel_launch(void* const* d_in, const int* in_sizes, int n_in,
                              void* d_out, int out_size, void* d_ws, size_t ws_size,
                              hipStream_t stream) {
  (void)in_sizes; (void)n_in; (void)out_size; (void)ws_size;
  const float* x         = (const float*)d_in[0];
  const int*   adj       = (const int*)d_in[1];
  const int*   adj1      = (const int*)d_in[2];
  const int*   use_adj   = (const int*)d_in[3];
  const float* edge_fea  = (const float*)d_in[4];
  const float* attn_bias = (const float*)d_in[5];
  const float* gn1_w = (const float*)d_in[6],  *gn1_b = (const float*)d_in[7],  *gn1_ms = (const float*)d_in[8];
  const float* gn2_w = (const float*)d_in[9],  *gn2_b = (const float*)d_in[10], *gn2_ms = (const float*)d_in[11];
  const float* ln1_w = (const float*)d_in[12], *ln1_b = (const float*)d_in[13];
  const float* ln2_w = (const float*)d_in[14], *ln2_b = (const float*)d_in[15];
  const float* Wq = (const float*)d_in[16], *bq = (const float*)d_in[17];
  const float* Wk = (const float*)d_in[18], *bk = (const float*)d_in[19];
  const float* Wv = (const float*)d_in[20], *bv = (const float*)d_in[21];
  const float* Wap = (const float*)d_in[22], *bap = (const float*)d_in[23];
  const float* Wep = (const float*)d_in[24], *bep = (const float*)d_in[25];
  const float* Wo = (const float*)d_in[26], *bo = (const float*)d_in[27];
  const float* Woe = (const float*)d_in[28];
  const float* W1 = (const float*)d_in[29], *b1 = (const float*)d_in[30];
  const float* W2 = (const float*)d_in[31], *b2 = (const float*)d_in[32];
  const float* W1e = (const float*)d_in[33], *b1e = (const float*)d_in[34];
  const float* W2e = (const float*)d_in[35], *b2e = (const float*)d_in[36];

  float* out_x = (float*)d_out;                       // [B,N,D]
  float* out_e = (float*)d_out + (long)B_ * N_ * D_;  // [B,N,N,E]

  // workspace bump allocator (256B aligned slabs)
  char* wp = (char*)d_ws;
  auto alloc = [&](size_t bytes) -> void* {
    void* p = (void*)wp; wp += (bytes + 255) & ~(size_t)255; return p;
  };
  _Float16* WqT = (_Float16*)alloc((size_t)D_ * D_ * 2);
  _Float16* WkT = (_Float16*)alloc((size_t)D_ * D_ * 2);
  _Float16* WvT = (_Float16*)alloc((size_t)D_ * D_ * 2);
  _Float16* WoT = (_Float16*)alloc((size_t)D_ * D_ * 2);
  _Float16* W1T = (_Float16*)alloc((size_t)D_ * FF_ * 2);
  _Float16* W2T = (_Float16*)alloc((size_t)FF_ * D_ * 2);
  _Float16* WoeT = (_Float16*)alloc((size_t)E_ * E_ * 2);
  _Float16* W1eT = (_Float16*)alloc((size_t)E_ * FF_ * 2);
  _Float16* W2eT = (_Float16*)alloc((size_t)FF_ * E_ * 2);
  _Float16* yh  = (_Float16*)alloc((size_t)B_ * N_ * D_ * 2);
  _Float16* qh  = (_Float16*)alloc((size_t)B_ * H_ * N_ * DK_ * 2);
  _Float16* kh  = (_Float16*)alloc((size_t)B_ * H_ * N_ * DK_ * 2);
  _Float16* vt  = (_Float16*)alloc((size_t)B_ * H_ * DK_ * N_ * 2);
  float*    scores = (float*)alloc((size_t)B_ * H_ * N_ * N_ * 4);
  _Float16* en_t   = (_Float16*)alloc((size_t)B_ * N_ * N_ * E_ * 2); // e_n then t
  float*    ebias  = (float*)alloc((size_t)B_ * H_ * N_ * N_ * 4);
  _Float16* P      = (_Float16*)alloc((size_t)B_ * H_ * N_ * N_ * 2);
  _Float16* attn_cat = (_Float16*)alloc((size_t)B_ * N_ * D_ * 2);
  float*    x1   = (float*)alloc((size_t)B_ * N_ * D_ * 4);
  _Float16* y2h  = (_Float16*)alloc((size_t)B_ * N_ * D_ * 2);
  float*    e1   = (float*)alloc((size_t)B_ * N_ * N_ * E_ * 4);
  _Float16* e1n  = (_Float16*)alloc((size_t)B_ * N_ * N_ * E_ * 2);

  auto cvt_t = [&](const float* s, _Float16* d, int K, int Nn) {
    int n = K * Nn;
    k_cvt_t<<<dim3((n + 255) / 256), dim3(256), 0, stream>>>(s, d, K, Nn);
  };
  cvt_t(Wq, WqT, D_, D_);   cvt_t(Wk, WkT, D_, D_);   cvt_t(Wv, WvT, D_, D_);
  cvt_t(Wo, WoT, D_, D_);   cvt_t(W1, W1T, D_, FF_);  cvt_t(W2, W2T, FF_, D_);
  cvt_t(Woe, WoeT, E_, E_); cvt_t(W1e, W1eT, E_, FF_); cvt_t(W2e, W2eT, FF_, E_);

  // node branch
  k_graph_norm<<<dim3(B_, D_ / 64), dim3(64), 0, stream>>>(x, gn1_w, gn1_b, gn1_ms, yh);
  k_qkv<<<dim3(B_ * N_ / 16, D_ / 16, 3), dim3(32), 0, stream>>>(
      yh, WqT, bq, WkT, bk, WvT, bv, qh, kh, vt);
  k_scores<<<dim3(N_ / 16, N_ / 16, B_ * H_), dim3(32), 0, stream>>>(
      qh, kh, adj, use_adj, scores);

  // edge LN + edge_bias
  k_edge_ln<<<dim3(B_ * N_ * N_ / 8), dim3(256), 0, stream>>>(
      edge_fea, ln1_w, ln1_b, en_t, Wep, bep, ebias);

  // dual softmax (global + local) and attention output
  k_softmax_dual<<<dim3(B_ * H_ * N_), dim3(256), 0, stream>>>(
      scores, attn_bias, ebias, adj1, P);
  k_attn_v<<<dim3(N_ / 16, B_ * H_), dim3(32), 0, stream>>>(P, vt, attn_cat);
  k_proj_res<<<dim3(B_ * N_ / 16, D_ / 16), dim3(32), 0, stream>>>(
      attn_cat, WoT, bo, x, x1);

  // node FFN (fused, hidden in LDS): out_x = x1 + FFN(graphnorm(x1))
  k_graph_norm<<<dim3(B_, D_ / 64), dim3(64), 0, stream>>>(x1, gn2_w, gn2_b, gn2_ms, y2h);
  k_fused_ffn<<<dim3(B_ * N_ / 16), dim3(32), 0, stream>>>(
      y2h, D_, W1T, b1, W2T, b2, D_, x1, out_x);

  // edge branch: t = e_n + x_proj (in place), e1 = edge_fea + t@Woe
  k_edge_fuse<<<dim3(B_ * N_ * N_ / 8), dim3(256), 0, stream>>>(
      scores, Wap, bap, adj1, en_t);
  k_edge_woe<<<dim3(B_ * N_ * N_ / 16), dim3(32), 0, stream>>>(
      en_t, WoeT, edge_fea, e1);
  k_edge_ln<<<dim3(B_ * N_ * N_ / 8), dim3(256), 0, stream>>>(
      e1, ln2_w, ln2_b, e1n, nullptr, nullptr, nullptr);

  // dominant edge FFN: TDM-staged transposed weights, 4 waves/block
  k_edge_ffn<<<dim3(B_ * N_ * N_ / 64), dim3(128), 0, stream>>>(
      e1n, W1eT, b1e, W2eT, b2e, e1, out_e);
}
